// Attention_80092550136278
// MI455X (gfx1250) — matrix operand order repaired and checked by reading, not compiled, not run on hardware
//
#include <hip/hip_runtime.h>

// Problem dims (fixed by reference)
#define HH   8
#define DD   32
#define QQ   2048
#define KVK  2048
#define CQd  256
#define CMd  256
#define COd  256

// bf16 WMMA fragment types (wave32, 16x16x32)
typedef __attribute__((ext_vector_type(16))) __bf16 v16bf;
typedef __attribute__((ext_vector_type(8)))  float  v8f;
typedef __attribute__((ext_vector_type(4)))  unsigned int u32x4;
typedef __attribute__((ext_vector_type(8)))  int    i32x8;
typedef __attribute__((ext_vector_type(4)))  int    i32x4;

union FragAB { v16bf v; unsigned int u[8]; };
union FragC  { v8f   v; float f[8]; };

__device__ __forceinline__ unsigned short f2bf(float x) {
  unsigned int u = __float_as_uint(x);
  u += 0x7FFFu + ((u >> 16) & 1u);   // round-to-nearest-even
  return (unsigned short)(u >> 16);
}

// K-offset of VGPR r for the 16-bit A fragment (contiguous k = 16*hi + 2*r, matching B)
__device__ __forceinline__ int a_koff(int r, int hi) {
  return 2 * r + 16 * hi;
}

// ---- Tensor Data Mover: 2-D f32 tile (rows x cols) -> LDS (packed) ----
// D# per CDNA5 ISA §8.3/8.4: group0 = {flags, lds_addr, global_addr, type=2},
// group1 = {data_size=4B, tensor dims/strides, tile dims}. Groups 2/3 unused (2-D).
__device__ __forceinline__ void tdm_load_f32_tile(const float* gsrc, unsigned int lds_off,
                                                  int rows, int cols, int row_stride_elems)
{
  unsigned long long ga = (unsigned long long)(size_t)gsrc;
  u32x4 g0;
  g0[0] = 1u;                                       // count=1, user descriptor
  g0[1] = lds_off;                                  // lds_addr (bytes)
  g0[2] = (unsigned int)(ga & 0xFFFFFFFFull);       // global_addr[31:0]
  g0[3] = (unsigned int)((ga >> 32) & 0x1FFFFFFull) // global_addr[56:32]
          | (2u << 30);                             // type = 2 ("image")
  i32x8 g1;
  g1[0] = (2 << 16);                                          // wg_mask=0, data_size=4B
  g1[1] = (row_stride_elems & 0xFFFF) << 16;                  // tensor_dim0[15:0]
  g1[2] = ((row_stride_elems >> 16) & 0xFFFF)                 // tensor_dim0[31:16]
          | ((rows & 0xFFFF) << 16);                          // tensor_dim1[15:0]
  g1[3] = ((rows >> 16) & 0xFFFF)                             // tensor_dim1[31:16]
          | ((cols & 0xFFFF) << 16);                          // tile_dim0
  g1[4] = rows & 0xFFFF;                                      // tile_dim1 (tile_dim2=0)
  g1[5] = row_stride_elems;                                   // tensor_dim0_stride[31:0]
  g1[6] = 0;
  g1[7] = 0;
  i32x4 z4 = {0, 0, 0, 0};
#if defined(__clang_major__) && (__clang_major__ >= 23)
  i32x8 z8 = {0, 0, 0, 0, 0, 0, 0, 0};
  __builtin_amdgcn_tensor_load_to_lds(g0, g1, z4, z4, z8, 0);
#else
  __builtin_amdgcn_tensor_load_to_lds(g0, g1, z4, z4, 0);
#endif
}

#define MODE_QK   0   // bf16 out[(h*S + m)*32 + d] * scale
#define MODE_VT   1   // bf16 out[(h*32 + d)*S + m]   (transposed per head)
#define MODE_GATE 2   // f32  out[m*N + n] = sigmoid(acc + bias[n])
#define MODE_OUT  3   // f32  out[m*N + n] = acc + bias[n]

// C[M,N] = A[M,K] @ (transW ? W^T : W), bf16 WMMA, f32 accumulate.
// Block tile 128x64, 256 threads = 8 waves, each wave: 16 rows x 64 cols = 4 WMMA tiles.
__global__ __launch_bounds__(256)
void proj_gemm_kernel(const float* __restrict__ A, const float* __restrict__ W,
                      const float* __restrict__ bias, float scale,
                      int M, int N, int Kd, int transW, int mode, int seqS,
                      unsigned short* __restrict__ obf, float* __restrict__ of32)
{
  __shared__ unsigned short As[128 * 34];   // padded row stride
  __shared__ unsigned short Bs[64 * 34];    // B^T tile: [n][k], contiguous in k
  const int tid = threadIdx.x;
  const int wv = tid >> 5, lane = tid & 31, l15 = lane & 15, hi = lane >> 4;
  const int m0 = blockIdx.x * 128, n0 = blockIdx.y * 64;

  FragC acc[4];
  #pragma unroll
  for (int t = 0; t < 4; ++t)
    #pragma unroll
    for (int r = 0; r < 8; ++r) acc[t].f[r] = 0.f;

  for (int k0 = 0; k0 < Kd; k0 += 32) {
    // Stage A tile (128x32), f32 -> bf16 fused
    #pragma unroll
    for (int i = 0; i < 16; ++i) {
      int idx = i * 256 + tid;
      int row = idx >> 5, col = idx & 31;
      As[row * 34 + col] = f2bf(A[(size_t)(m0 + row) * Kd + k0 + col]);
    }
    // Stage B^T tile (64x32)
    #pragma unroll
    for (int i = 0; i < 8; ++i) {
      int idx = i * 256 + tid;
      int n = idx >> 5, k = idx & 31;
      float wval = transW ? W[(size_t)(n0 + n) * Kd + (k0 + k)]
                          : W[(size_t)(k0 + k) * N + (n0 + n)];
      Bs[n * 34 + k] = f2bf(wval);
    }
    __syncthreads();

    FragAB af;
    #pragma unroll
    for (int r = 0; r < 8; ++r)
      af.u[r] = *(const unsigned int*)&As[(wv * 16 + l15) * 34 + a_koff(r, hi)];

    #pragma unroll
    for (int t = 0; t < 4; ++t) {
      FragAB bfr;
      #pragma unroll
      for (int r = 0; r < 8; ++r)
        bfr.u[r] = *(const unsigned int*)&Bs[(t * 16 + l15) * 34 + 2 * r + 16 * hi];
      acc[t].v = __builtin_amdgcn_wmma_f32_16x16x32_bf16(
          false, af.v, false, bfr.v, (short)0, acc[t].v, false, false);
    }
    __syncthreads();
  }

  // Epilogue
  #pragma unroll
  for (int t = 0; t < 4; ++t) {
    #pragma unroll
    for (int r = 0; r < 8; ++r) {
      int gm = m0 + wv * 16 + r + 8 * hi;
      int gn = n0 + t * 16 + l15;
      float val = acc[t].f[r] * scale;
      if (mode == MODE_QK) {
        int h = gn >> 5, d = gn & 31;
        obf[((size_t)h * seqS + gm) * 32 + d] = f2bf(val);
      } else if (mode == MODE_VT) {
        int h = gn >> 5, d = gn & 31;
        obf[(size_t)(h * 32 + d) * seqS + gm] = f2bf(val);
      } else if (mode == MODE_GATE) {
        float x = val + bias[gn];
        of32[(size_t)gm * N + gn] = 1.f / (1.f + __expf(-x));
      } else {
        of32[(size_t)gm * N + gn] = val + bias[gn];
      }
    }
  }
}

// Flash attention; pair_bias (the dominant 134 MB HBM stream) is staged into LDS
// by the Tensor Data Mover, double-buffered, overlapping the WMMA/softmax work.
// Grid: (Q/64, H). 128 threads = 4 waves; each wave owns 16 query rows.
__global__ __launch_bounds__(128)
void flash_attn_kernel(const unsigned short* __restrict__ qb,   // [h][Q][32] bf16 (pre-scaled)
                       const unsigned short* __restrict__ kb,   // [h][K][32] bf16
                       const unsigned short* __restrict__ vt,   // [h][32][K] bf16 (transposed)
                       const float* __restrict__ pb,            // [h][Q][K] f32
                       const float* __restrict__ gate,          // [Q][256] f32
                       float* __restrict__ y)                   // [Q][256] f32
{
  __shared__ float PB[2][64 * 64];           // TDM-staged pair_bias tiles (double buffer)
  __shared__ unsigned short P[4][16 * 68];   // wave-private P tiles (bf16)
  const int tid = threadIdx.x;
  const int wv = tid >> 5, lane = tid & 31, l15 = lane & 15, hi = lane >> 4;
  const int h = blockIdx.y;
  const int qb0 = blockIdx.x * 64;
  const int qbase = qb0 + wv * 16;
  const float L2E = 1.44269504088896f;

  // q A-fragment: D=32 == WMMA K, so a single fragment covers all of D
  FragAB qf;
  #pragma unroll
  for (int r = 0; r < 8; ++r)
    qf.u[r] = *(const unsigned int*)&qb[(size_t)(h * QQ + qbase + l15) * 32 + a_koff(r, hi)];

  FragC ctx[2];
  #pragma unroll
  for (int t = 0; t < 2; ++t)
    #pragma unroll
    for (int r = 0; r < 8; ++r) ctx[t].f[r] = 0.f;
  float mrow[8], lrow[8];
  #pragma unroll
  for (int r = 0; r < 8; ++r) { mrow[r] = -1e30f; lrow[r] = 0.f; }

  unsigned short* Pw = &P[wv][0];
  const float* pb_base = pb + (size_t)h * QQ * KVK + (size_t)qb0 * KVK;
  const unsigned int pb_lds0 = (unsigned int)(size_t)(void*)&PB[0][0];
  const unsigned int pb_lds1 = (unsigned int)(size_t)(void*)&PB[1][0];

  if (wv == 0)   // TDM: prime the pipeline with tile 0
    tdm_load_f32_tile(pb_base, pb_lds0, 64, 64, KVK);

  int it = 0;
  for (int kc = 0; kc < KVK; kc += 64, ++it) {
    if (wv == 0)
      __builtin_amdgcn_s_wait_tensorcnt(0);   // tile `it` has landed in LDS
    __syncthreads();                          // publish buffer to all waves
    if (wv == 0 && kc + 64 < KVK)             // issue tile it+1 into the other buffer
      tdm_load_f32_tile(pb_base + (kc + 64), (it & 1) ? pb_lds0 : pb_lds1, 64, 64, KVK);

    const float* pbt = &PB[it & 1][0];

    // S = q @ k^T for this 16x64 strip: 4 WMMA (K=32=D in one shot)
    FragC S[4];
    #pragma unroll
    for (int t = 0; t < 4; ++t) {
      FragAB kf;
      #pragma unroll
      for (int r = 0; r < 8; ++r)
        kf.u[r] = *(const unsigned int*)
            &kb[(size_t)(h * KVK + kc + t * 16 + l15) * 32 + 2 * r + 16 * hi];
      FragC z;
      #pragma unroll
      for (int r = 0; r < 8; ++r) z.f[r] = 0.f;
      S[t].v = __builtin_amdgcn_wmma_f32_16x16x32_bf16(
          false, qf.v, false, kf.v, (short)0, z.v, false, false);
    }

    // + pair_bias (from LDS; lane==column==bank -> conflict-free), running row max
    float rmax[8];
    #pragma unroll
    for (int r = 0; r < 8; ++r) rmax[r] = -1e30f;
    #pragma unroll
    for (int t = 0; t < 4; ++t)
      #pragma unroll
      for (int r = 0; r < 8; ++r) {
        float s = S[t].f[r] + pbt[(wv * 16 + r + 8 * hi) * 64 + t * 16 + l15];
        S[t].f[r] = s;
        rmax[r] = fmaxf(rmax[r], s);
      }
    #pragma unroll
    for (int msk = 8; msk >= 1; msk >>= 1)
      #pragma unroll
      for (int r = 0; r < 8; ++r)
        rmax[r] = fmaxf(rmax[r], __shfl_xor(rmax[r], msk, 32));

    float corr[8], rsum[8];
    #pragma unroll
    for (int r = 0; r < 8; ++r) {
      float mn = fmaxf(mrow[r], rmax[r]);
      corr[r] = exp2f((mrow[r] - mn) * L2E);
      mrow[r] = mn;
      rsum[r] = 0.f;
    }
    #pragma unroll
    for (int t = 0; t < 4; ++t)
      #pragma unroll
      for (int r = 0; r < 8; ++r) {
        float p = exp2f((S[t].f[r] - mrow[r]) * L2E);
        rsum[r] += p;
        Pw[(r + 8 * hi) * 68 + t * 16 + l15] = f2bf(p);   // wave-private LDS
      }
    #pragma unroll
    for (int msk = 8; msk >= 1; msk >>= 1)
      #pragma unroll
      for (int r = 0; r < 8; ++r)
        rsum[r] += __shfl_xor(rsum[r], msk, 32);
    #pragma unroll
    for (int r = 0; r < 8; ++r) {
      lrow[r] = lrow[r] * corr[r] + rsum[r];
      ctx[0].f[r] *= corr[r];
      ctx[1].f[r] *= corr[r];
    }

    // ctx += P @ V : 2 K-steps x 2 D-tiles = 4 WMMA
    #pragma unroll
    for (int ks = 0; ks < 2; ++ks) {
      FragAB pf;
      #pragma unroll
      for (int r = 0; r < 8; ++r)
        pf.u[r] = *(const unsigned int*)&Pw[l15 * 68 + ks * 32 + a_koff(r, hi)];
      #pragma unroll
      for (int t = 0; t < 2; ++t) {
        FragAB vf;
        #pragma unroll
        for (int r = 0; r < 8; ++r)
          vf.u[r] = *(const unsigned int*)
              &vt[(size_t)(h * 32 + t * 16 + l15) * KVK + kc + ks * 32 + 2 * r + 16 * hi];
        ctx[t].v = __builtin_amdgcn_wmma_f32_16x16x32_bf16(
            false, pf.v, false, vf.v, (short)0, ctx[t].v, false, false);
      }
    }
    __syncthreads();   // all waves done with PB[it&1] before TDM reuses it
  }

  // y = (ctx / l) * gate, f32, laid out [Q][H*D] for the final GEMM
  #pragma unroll
  for (int t = 0; t < 2; ++t)
    #pragma unroll
    for (int r = 0; r < 8; ++r) {
      int qrow = qbase + r + 8 * hi;
      int col = h * 32 + t * 16 + l15;
      y[(size_t)qrow * (HH * DD) + col] =
          (ctx[t].f[r] / lrow[r]) * gate[(size_t)qrow * (HH * DD) + col];
    }
}

extern "C" void kernel_launch(void* const* d_in, const int* in_sizes, int n_in,
                              void* d_out, int out_size, void* d_ws, size_t ws_size,
                              hipStream_t stream) {
  (void)in_sizes; (void)n_in; (void)out_size; (void)ws_size;
  const float* q_data    = (const float*)d_in[0];
  const float* k_data    = (const float*)d_in[1];
  const float* v_data    = (const float*)d_in[2];
  const float* pair_bias = (const float*)d_in[3];
  const float* Wq        = (const float*)d_in[4];
  const float* Wk        = (const float*)d_in[5];
  const float* Wv        = (const float*)d_in[6];
  const float* Wg        = (const float*)d_in[7];
  const float* Wo        = (const float*)d_in[8];
  const float* gating_b  = (const float*)d_in[9];
  const float* o_bias    = (const float*)d_in[10];
  float* out = (float*)d_out;

  char* ws = (char*)d_ws;
  unsigned short* qb = (unsigned short*)(ws);                 // 1 MB bf16 [h][Q][32]
  unsigned short* kb = (unsigned short*)(ws + (1 << 20));     // 1 MB bf16 [h][K][32]
  unsigned short* vt = (unsigned short*)(ws + (2 << 20));     // 1 MB bf16 [h][32][K]
  float*          gate = (float*)(ws + (3 << 20));            // 2 MB f32 [Q][256]
  float*          y    = (float*)(ws + (5 << 20));            // 2 MB f32 [Q][256]

  const dim3 blk(256);
  const dim3 gproj(QQ / 128, 256 / 64);   // 16 x 4
  const float qscale = 0.17677669529663687f;  // 1/sqrt(D)

  proj_gemm_kernel<<<gproj, blk, 0, stream>>>(q_data, Wq, (const float*)nullptr, qscale,
      QQ, HH * DD, CQd, 0, MODE_QK, QQ, qb, (float*)nullptr);
  proj_gemm_kernel<<<gproj, blk, 0, stream>>>(k_data, Wk, (const float*)nullptr, 1.f,
      KVK, HH * DD, CMd, 0, MODE_QK, KVK, kb, (float*)nullptr);
  proj_gemm_kernel<<<gproj, blk, 0, stream>>>(v_data, Wv, (const float*)nullptr, 1.f,
      KVK, HH * DD, CMd, 0, MODE_VT, KVK, vt, (float*)nullptr);
  proj_gemm_kernel<<<gproj, blk, 0, stream>>>(q_data, Wg, gating_b, 1.f,
      QQ, HH * DD, CQd, 1, MODE_GATE, 0, (unsigned short*)nullptr, gate);

  flash_attn_kernel<<<dim3(QQ / 64, HH), dim3(128), 0, stream>>>(qb, kb, vt, pair_bias, gate, y);

  proj_gemm_kernel<<<gproj, blk, 0, stream>>>(y, Wo, o_bias, 1.f,
      QQ, COd, HH * DD, 1, MODE_OUT, 0, (unsigned short*)nullptr, out);
}
